// AdaptiveThresholdLIFNeuron_56470230008107
// MI455X (gfx1250) — compile-verified, matrix operand order checked by reading, not run on hardware
//
#include <hip/hip_runtime.h>

// ---------------------------------------------------------------------------
// Adaptive-threshold LIF scan for MI455X (gfx1250, wave32).
//
// Parallelization: 32 blocks x 256 threads, 16 neurons/thread (32*256*16 =
// 131072 = B*F neurons). Per timestep the only cross-neuron coupling is the
// scalar EMA of the spike rate, implemented as an exact integer grid
// reduction + spin barrier (32 resident blocks -> deadlock-safe, cheap).
//
// CDNA5 specifics:
//  - per-wave spike counting via V_WMMA_F32_16X16X32_F16 (ones x spikes):
//    one wave owns exactly 512 neurons = one 32x16 f16 B matrix; result is
//    exact because spikes are 0/1 and accumulation is f32.
//  - s_sleep in the grid-barrier spin loop.
// ---------------------------------------------------------------------------

typedef __attribute__((ext_vector_type(16))) _Float16 v16h;
typedef __attribute__((ext_vector_type(8)))  float    v8f;

#define NBLK 32
#define NTHR 256
#define NPT  16                        // neurons per thread
#define TOTAL_THREADS (NBLK * NTHR)    // 8192
#define N_NEURONS 131072               // 32 * 4096
#define T_STEPS 256
#define INV_N (1.0f / 131072.0f)

// float(np.exp(-dt/tau)) constants (f64-derived literals, rounded to f32):
#define K_ALPHA 0.9512294245007140f    // exp(-0.001/0.02)
#define K_BETA  0.8187307530779818f    // exp(-0.001/0.005)
#define K_GAMMA 0.9900498337491681f    // exp(-0.001/0.1)
#define K_1MG   0.0099501662508319f    // 1.0 - exp(-0.001/0.1) (in f64)

__device__ __forceinline__ float waveAllSum(float v) {
    v += __shfl_xor(v, 1,  32);
    v += __shfl_xor(v, 2,  32);
    v += __shfl_xor(v, 4,  32);
    v += __shfl_xor(v, 8,  32);
    v += __shfl_xor(v, 16, 32);
    return v;
}

__global__ void lif_init_kernel(unsigned int* __restrict__ ctrs) {
    // zero g_spk[256] and g_bar[256]
    int i = threadIdx.x;
    if (i < 512) ctrs[i] = 0u;
}

__global__ __launch_bounds__(NTHR, 1)
void lif_scan_kernel(const float* __restrict__ x,
                     const float* __restrict__ thScalePtr,
                     const float* __restrict__ adBiasPtr,
                     float* __restrict__ spikes,
                     float* __restrict__ emaOut,
                     unsigned int* __restrict__ g_spk,
                     unsigned int* __restrict__ g_bar,
                     float* __restrict__ pMem,
                     float* __restrict__ pTh,
                     float* __restrict__ pAdapt)
{
    const int tid  = threadIdx.x;
    const int lane = tid & 31;
    const int wid  = tid >> 5;
    const int gtid = (int)blockIdx.x * NTHR + tid;

    __shared__ float sSpk[8];
    __shared__ float sMem[8];
    __shared__ float sTh[8];
    __shared__ unsigned int sCnt;

    const float thScale = thScalePtr[0];
    const float adBias  = adBiasPtr[0];

    float mem[NPT], syn[NPT], adp[NPT];
#pragma unroll
    for (int j = 0; j < NPT; ++j) { mem[j] = 0.f; syn[j] = 0.f; adp[j] = 0.f; }

    float ema = 0.f;

    v16h ones;
#pragma unroll
    for (int j = 0; j < 16; ++j) ones[j] = (_Float16)1.0f;

    const float4* __restrict__ xv = (const float4*)x;   // [neuron][64] float4

    for (int tg = 0; tg < T_STEPS / 4; ++tg) {
        // Stage 4 timesteps of input for all 16 neurons (contiguous in T).
        float4 xb[NPT];
#pragma unroll
        for (int j = 0; j < NPT; ++j)
            xb[j] = xv[(gtid + j * TOTAL_THREADS) * 64 + tg];

#pragma unroll
        for (int s = 0; s < 4; ++s) {
            const int t = tg * 4 + s;
            // homeostasis uses ema from BEFORE this step's update
            const float homeo = 0.1f * fmaxf(ema - 0.01f, 0.0f);
            const float thC   = 1.0f * thScale + homeo + adBias;

            float memSum = 0.f, thSum = 0.f;
            v16h spkh;
#pragma unroll
            for (int j = 0; j < NPT; ++j) {
                const float xin = (s == 0) ? xb[j].x :
                                  (s == 1) ? xb[j].y :
                                  (s == 2) ? xb[j].z : xb[j].w;
                syn[j] = K_BETA  * syn[j] + xin;
                mem[j] = K_ALPHA * mem[j] + syn[j];
                adp[j] = K_GAMMA * adp[j] + K_1MG * mem[j];
                const float th  = thC + 0.1f * adp[j];
                const float spk = (mem[j] >= th) ? 1.0f : 0.0f;
                mem[j] = mem[j] - th * spk;
                memSum += mem[j];
                thSum  += th;
                spkh[j] = (_Float16)spk;
                spikes[(gtid + j * TOTAL_THREADS) * T_STEPS + t] = spk;
            }

            // Exact per-wave spike count via the matrix unit:
            // D = ones(16x32) * spikes(32x16); every D element is a column
            // sum, so sum of VGPR0 over all 32 lanes = 2 * total (exact).
            v8f acc = {};
            acc = __builtin_amdgcn_wmma_f32_16x16x32_f16(
                false, ones, false, spkh, (short)0, acc, false, false);
            float waveSpk = waveAllSum(acc[0]) * 0.5f;

            memSum = waveAllSum(memSum);
            thSum  = waveAllSum(thSum);

            if (lane == 0) { sSpk[wid] = waveSpk; sMem[wid] = memSum; sTh[wid] = thSum; }
            __syncthreads();

            if (tid == 0) {
                float bs = 0.f, bm = 0.f, bt = 0.f;
#pragma unroll
                for (int w = 0; w < 8; ++w) { bs += sSpk[w]; bm += sMem[w]; bt += sTh[w]; }
                // deterministic partials (reduced in fixed order later)
                pMem[t * NBLK + blockIdx.x] = bm;
                pTh [t * NBLK + blockIdx.x] = bt;
                // exact integer spike count: order-independent atomic add
                const unsigned int c = (unsigned int)(bs + 0.5f);
                __hip_atomic_fetch_add(&g_spk[t], c, __ATOMIC_RELAXED,
                                       __HIP_MEMORY_SCOPE_AGENT);
                __hip_atomic_fetch_add(&g_bar[t], 1u, __ATOMIC_RELEASE,
                                       __HIP_MEMORY_SCOPE_AGENT);
                while (__hip_atomic_load(&g_bar[t], __ATOMIC_ACQUIRE,
                                         __HIP_MEMORY_SCOPE_AGENT) < NBLK) {
                    __builtin_amdgcn_s_sleep(1);
                }
                sCnt = __hip_atomic_load(&g_spk[t], __ATOMIC_RELAXED,
                                         __HIP_MEMORY_SCOPE_AGENT);
            }
            __syncthreads();

            // all threads update an identical private copy of ema
            ema = 0.99f * ema + 0.01f * ((float)sCnt * INV_N);
        }
    }

    // final adapt partial sum (deterministic two-stage reduction)
    float aSum = 0.f;
#pragma unroll
    for (int j = 0; j < NPT; ++j) aSum += adp[j];
    aSum = waveAllSum(aSum);
    if (lane == 0) sMem[wid] = aSum;
    __syncthreads();
    if (tid == 0) {
        float a = 0.f;
#pragma unroll
        for (int w = 0; w < 8; ++w) a += sMem[w];
        pAdapt[blockIdx.x] = a;
        if (blockIdx.x == 0) emaOut[0] = ema;
    }
}

__global__ void lif_finalize_kernel(const float* __restrict__ pMem,
                                    const float* __restrict__ pTh,
                                    const float* __restrict__ pAdapt,
                                    float* __restrict__ memTrace,
                                    float* __restrict__ thTrace,
                                    float* __restrict__ adaptMean)
{
    const int t = threadIdx.x;   // one thread per timestep, fixed-order sums
    float m = 0.f, h = 0.f;
#pragma unroll 4
    for (int b = 0; b < NBLK; ++b) {
        m += pMem[t * NBLK + b];
        h += pTh [t * NBLK + b];
    }
    memTrace[t] = m * INV_N;
    thTrace[t]  = h * INV_N;
    if (t == 0) {
        float a = 0.f;
        for (int b = 0; b < NBLK; ++b) a += pAdapt[b];
        adaptMean[0] = a * INV_N;
    }
}

extern "C" void kernel_launch(void* const* d_in, const int* in_sizes, int n_in,
                              void* d_out, int out_size, void* d_ws, size_t ws_size,
                              hipStream_t stream)
{
    (void)in_sizes; (void)n_in; (void)out_size; (void)ws_size;

    const float* x       = (const float*)d_in[0];   // [B,F,T] f32
    const float* thScale = (const float*)d_in[1];   // [1]
    const float* adBias  = (const float*)d_in[2];   // [1]

    float* out = (float*)d_out;
    const size_t S = (size_t)N_NEURONS * T_STEPS;   // 33554432
    float* spikes    = out;
    float* memTrace  = out + S;
    float* thTrace   = out + S + T_STEPS;
    float* emaOut    = out + S + 2 * T_STEPS;
    float* adaptMean = out + S + 2 * T_STEPS + 1;

    unsigned int* g_spk = (unsigned int*)d_ws;          // [256]
    unsigned int* g_bar = g_spk + T_STEPS;              // [256]
    float* pMem   = (float*)(g_bar + T_STEPS);          // [256*32]
    float* pTh    = pMem + T_STEPS * NBLK;              // [256*32]
    float* pAdapt = pTh  + T_STEPS * NBLK;              // [32]

    lif_init_kernel<<<1, 512, 0, stream>>>(g_spk);
    lif_scan_kernel<<<NBLK, NTHR, 0, stream>>>(x, thScale, adBias, spikes, emaOut,
                                               g_spk, g_bar, pMem, pTh, pAdapt);
    lif_finalize_kernel<<<1, T_STEPS, 0, stream>>>(pMem, pTh, pAdapt,
                                                   memTrace, thTrace, adaptMean);
}
